// DEQBrain_20126216749541
// MI455X (gfx1250) — matrix operand order repaired and checked by reading, not compiled, not run on hardware
//
#include <hip/hip_runtime.h>
#include <hip/hip_bf16.h>
#include <math.h>

typedef __bf16 bf16;
typedef __attribute__((ext_vector_type(16))) __bf16 bf16x16;
typedef __attribute__((ext_vector_type(8)))  float   f32x8;

#define B_SZ 4
#define T_SZ 1024
#define C_SZ 768
#define H_SZ 12
#define DH   64
#define NTOK 4096          // B*T
#define NUM_ITERS 6
#define M_HIST 5

// ---------------------------------------------------------------------------
// CDNA5 async global->LDS copy (ASYNCcnt path), inline asm per ISA ch.10/15.18
// ---------------------------------------------------------------------------
__device__ __forceinline__ uint32_t lds_addr32(const void* p) {
  // generic LDS pointer: addr[63:32]=shared aperture, addr[31:0]=LDS address
  return (uint32_t)(uintptr_t)p;
}
__device__ __forceinline__ void async_b128(const void* gsrc, uint32_t ldsoff) {
  asm volatile("global_load_async_to_lds_b128 %0, %1, off"
               :
               : "v"(ldsoff), "v"((unsigned long long)(uintptr_t)gsrc)
               : "memory");
}
__device__ __forceinline__ void wait_async0() {
  asm volatile("s_wait_asynccnt 0" ::: "memory");
}

// ---------------------------------------------------------------------------
// WMMA helpers (CDNA5 16x16x32 bf16, f32 accumulate)
// ---------------------------------------------------------------------------
__device__ __forceinline__ f32x8 wmma_bf16(bf16x16 a, bf16x16 b, f32x8 c) {
  return __builtin_amdgcn_wmma_f32_16x16x32_bf16(
      /*neg_a=*/false, a, /*neg_b=*/false, b,
      /*c_mod=*/(short)0, c, /*reuse_a=*/false, /*reuse_b=*/false);
}

// A-matrix 16x32 bf16 fragment from a row-major LDS tile [16][ldk]
// ISA layout: V0..3 lanes0-15 K=0..7 (pairs), lanes16-31 K=8..15; V4..7 +16.
__device__ __forceinline__ bf16x16 load_a_frag(const bf16* tile, int lane, int ldk) {
  const int m  = lane & 15;
  const int hi = (lane >> 4) & 1;
  bf16x16 a;
#pragma unroll
  for (int v = 0; v < 8; ++v) {
    const int grp = v >> 2, vv = v & 3;
    const int k = grp * 16 + hi * 8 + 2 * vv;
    a[2 * v]     = tile[m * ldk + k];
    a[2 * v + 1] = tile[m * ldk + k + 1];
  }
  return a;
}

// B-matrix 32x16 bf16 fragment. tileT is B transposed, row-major [16 n][ldk k]
// ISA layout: V0..3 lanes0-15 K=0..7 (pairs), lanes16-31 K=8..15; V4..7 +16.
__device__ __forceinline__ bf16x16 load_b_frag(const bf16* tileT, int lane, int ldk) {
  const int n  = lane & 15;
  const int hi = (lane >> 4) & 1;
  bf16x16 b;
#pragma unroll
  for (int v = 0; v < 8; ++v) {
    const int k = (v >> 2) * 16 + hi * 8 + 2 * (v & 3);
    b[2 * v]     = tileT[n * ldk + k];
    b[2 * v + 1] = tileT[n * ldk + k + 1];
  }
  return b;
}

// ---------------------------------------------------------------------------
// fp32 -> bf16 convert (weights, once per launch)
// ---------------------------------------------------------------------------
__global__ __launch_bounds__(256) void cvt_bf16_kernel(const float* __restrict__ in,
                                                       bf16* __restrict__ out, int n) {
  int i = blockIdx.x * 256 + threadIdx.x;
  if (i < n) out[i] = (bf16)in[i];
}

// ---------------------------------------------------------------------------
// LayerNorm (one wave per token, C=768 -> 24 elems/lane), output bf16
// WITH_U: x = z + 0.1*emb + u  (LN1); else x = z (LN2)
// ---------------------------------------------------------------------------
template <int WITH_U>
__global__ __launch_bounds__(256) void ln_kernel(const float* __restrict__ zin,
                                                 const float* __restrict__ u,
                                                 const float* __restrict__ emb,
                                                 const float* __restrict__ w,
                                                 const float* __restrict__ bgain,
                                                 bf16* __restrict__ out) {
  const int tok  = blockIdx.x * 8 + (threadIdx.x >> 5);
  const int lane = threadIdx.x & 31;
  const size_t base = (size_t)tok * C_SZ;
  float v[24];
  float s = 0.f;
#pragma unroll
  for (int j = 0; j < 24; ++j) {
    const int c = lane + 32 * j;
    float x = zin[base + c];
    if constexpr (WITH_U) x += 0.1f * emb[c] + u[base + c];
    v[j] = x;
    s += x;
  }
#pragma unroll
  for (int m = 16; m >= 1; m >>= 1) s += __shfl_xor(s, m, 32);
  const float mu = s * (1.0f / (float)C_SZ);
  float vs = 0.f;
#pragma unroll
  for (int j = 0; j < 24; ++j) {
    const float d = v[j] - mu;
    v[j] = d;
    vs += d * d;
  }
#pragma unroll
  for (int m = 16; m >= 1; m >>= 1) vs += __shfl_xor(vs, m, 32);
  const float inv = rsqrtf(vs * (1.0f / (float)C_SZ) + 1e-5f);
#pragma unroll
  for (int j = 0; j < 24; ++j) {
    const int c = lane + 32 * j;
    out[base + c] = (bf16)(v[j] * inv * w[c] + bgain[c]);
  }
}

// ---------------------------------------------------------------------------
// Generic bf16 GEMM: C[M,N] = A[M,K] x Bt[N,K]^T  (+ bias, fused epilogues)
// Block 256 thr = 8 waves; block tile 128x128; wave tile 64x32; K step 32.
// Double-buffered LDS slabs staged with global_load_async_to_lds_b128.
// MODE 1: QKV pack -> qb/kb/vb bf16 [BH,T,DH], q scaled by 0.125
// MODE 2: outF = acc + bias + resid                  (out-proj residual)
// MODE 3: outB = bf16(gelu(acc + bias))              (MLP fc1)
// MODE 4: outF = acc + bias + resid - sub            (MLP fc2 -> delta)
// ---------------------------------------------------------------------------
template <int MODE>
__global__ __launch_bounds__(256) void gemm_bf16_kernel(
    const bf16* __restrict__ A, const bf16* __restrict__ Bt,
    const float* __restrict__ bias, float* __restrict__ outF,
    bf16* __restrict__ outB, const float* __restrict__ resid,
    const float* __restrict__ sub, bf16* __restrict__ qb, bf16* __restrict__ kb,
    bf16* __restrict__ vb, int M, int N, int K) {
  __shared__ bf16 As[2][128 * 32];
  __shared__ bf16 Bs[2][128 * 32];
  const int tid  = threadIdx.x;
  const int lane = tid & 31;
  const int wid  = tid >> 5;
  const int m0 = blockIdx.y * 128;
  const int n0 = blockIdx.x * 128;
  const int wm = (wid >> 2) * 64;  // 0 / 64
  const int wn = (wid & 3) * 32;   // 0 / 32 / 64 / 96

  f32x8 acc[4][2] = {};

  const int lr = tid >> 1;         // 0..127 slab row
  const int lc = (tid & 1) * 16;   // 16-element chunk (32 bytes)

  // issue async stage of one k-slab pair into buffer `buf`
  auto stage = [&](int buf, int k0) {
    const bf16* ag = A + (size_t)(m0 + lr) * K + k0 + lc;
    const bf16* bg = Bt + (size_t)(n0 + lr) * K + k0 + lc;
    const uint32_t ao = lds_addr32(&As[buf][lr * 32 + lc]);
    const uint32_t bo = lds_addr32(&Bs[buf][lr * 32 + lc]);
    async_b128(ag, ao);
    async_b128(ag + 8, ao + 16);
    async_b128(bg, bo);
    async_b128(bg + 8, bo + 16);
  };

  const int nk = K >> 5;
  stage(0, 0);
  for (int ki = 0; ki < nk; ++ki) {
    wait_async0();       // this wave's slab loads (buffer ki&1) are in LDS
    __syncthreads();     // all waves loaded; all waves done reading buf (ki+1)&1
    if (ki + 1 < nk) stage((ki + 1) & 1, (ki + 1) << 5);

    const bf16* Asb = As[ki & 1];
    const bf16* Bsb = Bs[ki & 1];
    bf16x16 bf0 = load_b_frag(Bsb + (wn + 0) * 32, lane, 32);
    bf16x16 bf1 = load_b_frag(Bsb + (wn + 16) * 32, lane, 32);
#pragma unroll
    for (int mt = 0; mt < 4; ++mt) {
      bf16x16 af = load_a_frag(Asb + (wm + mt * 16) * 32, lane, 32);
      acc[mt][0] = wmma_bf16(af, bf0, acc[mt][0]);
      acc[mt][1] = wmma_bf16(af, bf1, acc[mt][1]);
    }
    __syncthreads();     // keep reads of buf ki&1 ahead of its re-stage
  }

  // epilogue: C layout -> row = i + 8*(lane>=16), col = lane&15
  const int cn = lane & 15;
  const int rh = (lane >> 4) * 8;
#pragma unroll
  for (int mt = 0; mt < 4; ++mt) {
#pragma unroll
    for (int nt = 0; nt < 2; ++nt) {
#pragma unroll
      for (int i = 0; i < 8; ++i) {
        const int m = m0 + wm + mt * 16 + i + rh;
        const int n = n0 + wn + nt * 16 + cn;
        const float v = acc[mt][nt][i] + bias[n];
        if constexpr (MODE == 1) {
          const int which = n / C_SZ;
          const int c = n % C_SZ;
          const int h = c / DH, d = c % DH;
          const int b = m / T_SZ, t = m % T_SZ;
          const size_t dst = ((size_t)(b * H_SZ + h) * T_SZ + t) * DH + d;
          if (which == 0)
            qb[dst] = (bf16)(v * 0.125f);  // 1/sqrt(dh)
          else if (which == 1)
            kb[dst] = (bf16)v;
          else
            vb[dst] = (bf16)v;
        } else if constexpr (MODE == 2) {
          const size_t idx = (size_t)m * N + n;
          outF[idx] = v + resid[idx];
        } else if constexpr (MODE == 3) {
          const float g = 0.5f * v * (1.0f + erff(v * 0.70710678118654752f));
          outB[(size_t)m * N + n] = (bf16)g;
        } else {  // MODE == 4
          const size_t idx = (size_t)m * N + n;
          outF[idx] = v + resid[idx] - sub[idx];
        }
      }
    }
  }
}

// ---------------------------------------------------------------------------
// Flash attention: grid (16 q-tiles, 48 batch*head), block 128 thr = 4 waves.
// Each wave: 16 q rows x full dh=64; online softmax; P round-trips via LDS
// to convert C-layout f32 -> A-layout bf16. Q/K slabs staged asynchronously.
// ---------------------------------------------------------------------------
__global__ __launch_bounds__(128) void flash_attn_kernel(
    const bf16* __restrict__ qb, const bf16* __restrict__ kb,
    const bf16* __restrict__ vb, bf16* __restrict__ outb) {
  __shared__ bf16 Qs[64 * 64];
  __shared__ bf16 Ks[64 * 64];
  __shared__ bf16 Vt[64 * 64];       // V transposed: [d][t]
  __shared__ bf16 Ps[4 * 16 * 64];   // per-wave P staging
  const int tid  = threadIdx.x;
  const int lane = tid & 31;
  const int wid  = tid >> 5;
  const int qt = blockIdx.x;
  const int bh = blockIdx.y;
  const int b = bh / H_SZ, h = bh % H_SZ;

  // stage Q slab (contiguous 64x64 bf16 = 512 x b128) via async copy
  {
    const bf16* qsrc = qb + ((size_t)bh * T_SZ + qt * 64) * DH;
    for (int i = tid; i < 512; i += 128)
      async_b128(qsrc + i * 8, lds_addr32(Qs + i * 8));
    wait_async0();
  }
  __syncthreads();

  const int qr = wid * 16;
  const bf16x16 aq0 = load_a_frag(Qs + qr * 64, lane, 64);
  const bf16x16 aq1 = load_a_frag(Qs + qr * 64 + 32, lane, 64);

  f32x8 o[4] = {};
  float mrun[8], lrun[8];
#pragma unroll
  for (int i = 0; i < 8; ++i) {
    mrun[i] = -1e30f;
    lrun[i] = 0.f;
  }
  bf16* Pw = Ps + wid * 16 * 64;

  for (int kt = 0; kt < 16; ++kt) {
    __syncthreads();
    {
      const bf16* ksrc = kb + ((size_t)bh * T_SZ + kt * 64) * DH;
      for (int i = tid; i < 512; i += 128)
        async_b128(ksrc + i * 8, lds_addr32(Ks + i * 8));
      const bf16* vsrc = vb + ((size_t)bh * T_SZ + kt * 64) * DH;
      for (int i = tid; i < 4096; i += 128) {
        const int t = i >> 6, dd = i & 63;
        Vt[dd * 64 + t] = vsrc[i];   // transposing scatter (manual)
      }
      wait_async0();
    }
    __syncthreads();

    // S = Q * K^T (16 x 64), 8 WMMAs
    f32x8 sc[4] = {};
#pragma unroll
    for (int nt = 0; nt < 4; ++nt) {
      bf16x16 bk0 = load_b_frag(Ks + nt * 16 * 64, lane, 64);
      bf16x16 bk1 = load_b_frag(Ks + nt * 16 * 64 + 32, lane, 64);
      sc[nt] = wmma_bf16(aq0, bk0, sc[nt]);
      sc[nt] = wmma_bf16(aq1, bk1, sc[nt]);
    }

    // online softmax (row-wise over 64 cols: 4 tiles x 16 lanes of our half)
#pragma unroll
    for (int i = 0; i < 8; ++i) {
      float mx = fmaxf(fmaxf(sc[0][i], sc[1][i]), fmaxf(sc[2][i], sc[3][i]));
#pragma unroll
      for (int m = 8; m >= 1; m >>= 1) mx = fmaxf(mx, __shfl_xor(mx, m, 32));
      const float mnew = fmaxf(mrun[i], mx);
      const float scale = __expf(mrun[i] - mnew);
      float rs = 0.f;
#pragma unroll
      for (int nt = 0; nt < 4; ++nt) {
        const float p = __expf(sc[nt][i] - mnew);
        sc[nt][i] = p;
        rs += p;
      }
#pragma unroll
      for (int m = 8; m >= 1; m >>= 1) rs += __shfl_xor(rs, m, 32);
      lrun[i] = lrun[i] * scale + rs;
#pragma unroll
      for (int nt = 0; nt < 4; ++nt) o[nt][i] *= scale;
      mrun[i] = mnew;
    }

    // P (f32 C-layout) -> bf16 A-layout via per-wave LDS staging
    {
      const int cn = lane & 15, rh = (lane >> 4) * 8;
#pragma unroll
      for (int nt = 0; nt < 4; ++nt)
#pragma unroll
        for (int i = 0; i < 8; ++i)
          Pw[(i + rh) * 64 + nt * 16 + cn] = (bf16)sc[nt][i];
    }
    bf16x16 ap0 = load_a_frag(Pw, lane, 64);
    bf16x16 ap1 = load_a_frag(Pw + 32, lane, 64);

    // O += P * V (k-dim = 64 key tokens), 8 WMMAs
#pragma unroll
    for (int nt = 0; nt < 4; ++nt) {
      bf16x16 bv0 = load_b_frag(Vt + nt * 16 * 64, lane, 64);
      bf16x16 bv1 = load_b_frag(Vt + nt * 16 * 64 + 32, lane, 64);
      o[nt] = wmma_bf16(ap0, bv0, o[nt]);
      o[nt] = wmma_bf16(ap1, bv1, o[nt]);
    }
  }

  // epilogue: normalize and scatter to token-major [tok, h*64+d] bf16
  const int cn = lane & 15, rh = (lane >> 4) * 8;
#pragma unroll
  for (int nt = 0; nt < 4; ++nt) {
#pragma unroll
    for (int i = 0; i < 8; ++i) {
      const int t = qt * 64 + qr + i + rh;
      const size_t tok = (size_t)b * T_SZ + t;
      const int d = nt * 16 + cn;
      outb[tok * C_SZ + h * DH + d] = (bf16)(o[nt][i] / lrun[i]);
    }
  }
}

// ---------------------------------------------------------------------------
// Anderson acceleration update: one wave per token.
// res = Fs[newest]; dF[j] = Fs[old_j] - res; G = dF dF^T + 1e-6 I;
// alpha = G^-1 (dF r); z += r - dF^T alpha.  (beta = 1)
// ---------------------------------------------------------------------------
__global__ __launch_bounds__(256) void anderson_kernel(
    float* __restrict__ z, const float* __restrict__ Fs, int newest, int K,
    int o0, int o1, int o2, int o3) {
  const int tok  = blockIdx.x * 8 + (threadIdx.x >> 5);
  const int lane = threadIdx.x & 31;
  const size_t base = (size_t)tok * C_SZ;
  const size_t SL = (size_t)NTOK * C_SZ;

  float r[24];
#pragma unroll
  for (int j = 0; j < 24; ++j)
    r[j] = Fs[(size_t)newest * SL + base + lane + 32 * j];

  if (K == 0) {
#pragma unroll
    for (int j = 0; j < 24; ++j) z[base + lane + 32 * j] += r[j];
    return;
  }

  const int olds[4] = {o0, o1, o2, o3};
  float dF[4][24];
  for (int a = 0; a < K; ++a)
#pragma unroll
    for (int j = 0; j < 24; ++j)
      dF[a][j] = Fs[(size_t)olds[a] * SL + base + lane + 32 * j] - r[j];

  float G[4][4], bv[4], alpha[4] = {0.f, 0.f, 0.f, 0.f};
  for (int a = 0; a < K; ++a) {
    for (int c = a; c < K; ++c) {
      float s = 0.f;
#pragma unroll
      for (int j = 0; j < 24; ++j) s += dF[a][j] * dF[c][j];
#pragma unroll
      for (int m = 16; m >= 1; m >>= 1) s += __shfl_xor(s, m, 32);
      G[a][c] = s;
      G[c][a] = s;
    }
    G[a][a] += 1e-6f;
    float s = 0.f;
#pragma unroll
    for (int j = 0; j < 24; ++j) s += dF[a][j] * r[j];
#pragma unroll
    for (int m = 16; m >= 1; m >>= 1) s += __shfl_xor(s, m, 32);
    bv[a] = s;
  }

  // redundant per-lane Gaussian elimination (K <= 4, SPD + reg)
  for (int p = 0; p < K; ++p) {
    const float piv = 1.f / G[p][p];
    for (int q = p + 1; q < K; ++q) {
      const float f = G[q][p] * piv;
      for (int c = p; c < K; ++c) G[q][c] -= f * G[p][c];
      bv[q] -= f * bv[p];
    }
  }
  for (int p = K - 1; p >= 0; --p) {
    float sAcc = bv[p];
    for (int c = p + 1; c < K; ++c) sAcc -= G[p][c] * alpha[c];
    alpha[p] = sAcc / G[p][p];
  }

#pragma unroll
  for (int j = 0; j < 24; ++j) {
    float d = r[j];
    for (int a = 0; a < K; ++a) d -= alpha[a] * dF[a][j];
    z[base + lane + 32 * j] += d;
  }
}

// ---------------------------------------------------------------------------
// host orchestration
// ---------------------------------------------------------------------------
static inline int imin_host(int a, int b) { return a < b ? a : b; }

extern "C" void kernel_launch(void* const* d_in, const int* in_sizes, int n_in,
                              void* d_out, int out_size, void* d_ws,
                              size_t ws_size, hipStream_t stream) {
  (void)in_sizes; (void)n_in; (void)out_size; (void)ws_size;
  const float* u          = (const float*)d_in[0];
  const float* iter_emb   = (const float*)d_in[1];
  const float* ln1_w      = (const float*)d_in[2];
  const float* ln1_b      = (const float*)d_in[3];
  const float* in_proj_w  = (const float*)d_in[4];
  const float* in_proj_b  = (const float*)d_in[5];
  const float* out_proj_w = (const float*)d_in[6];
  const float* out_proj_b = (const float*)d_in[7];
  const float* ln2_w      = (const float*)d_in[8];
  const float* ln2_b      = (const float*)d_in[9];
  const float* mlp_w1     = (const float*)d_in[10];
  const float* mlp_b1     = (const float*)d_in[11];
  const float* mlp_w2     = (const float*)d_in[12];
  const float* mlp_b2     = (const float*)d_in[13];

  // ---- workspace carve ----
  uint8_t* p = (uint8_t*)d_ws;
  auto carve = [&](size_t bytes) {
    void* r = (void*)p;
    p += (bytes + 255) & ~(size_t)255;
    return r;
  };
  const size_t TOKC = (size_t)NTOK * C_SZ;
  float* zbuf   = (float*)carve(TOKC * 4);
  float* z_attn = (float*)carve(TOKC * 4);
  float* FsBuf  = (float*)carve(TOKC * 4 * M_HIST);
  bf16* x_bf    = (bf16*)carve(TOKC * 2);
  bf16* h_bf    = (bf16*)carve(TOKC * 2);
  bf16* attn_bf = (bf16*)carve(TOKC * 2);
  bf16* g_bf    = (bf16*)carve((size_t)NTOK * 4 * C_SZ * 2);
  bf16* qb      = (bf16*)carve(TOKC * 2);
  bf16* kb      = (bf16*)carve(TOKC * 2);
  bf16* vb      = (bf16*)carve(TOKC * 2);
  bf16* wqkv_bf = (bf16*)carve((size_t)3 * C_SZ * C_SZ * 2);
  bf16* wout_bf = (bf16*)carve((size_t)C_SZ * C_SZ * 2);
  bf16* w1_bf   = (bf16*)carve((size_t)4 * C_SZ * C_SZ * 2);
  bf16* w2_bf   = (bf16*)carve((size_t)4 * C_SZ * C_SZ * 2);

  // ---- weight conversion (weights are already [N,K] = Bt layout) ----
  {
    const int n1 = 3 * C_SZ * C_SZ, n2 = C_SZ * C_SZ, n3 = 4 * C_SZ * C_SZ;
    cvt_bf16_kernel<<<(n1 + 255) / 256, 256, 0, stream>>>(in_proj_w, wqkv_bf, n1);
    cvt_bf16_kernel<<<(n2 + 255) / 256, 256, 0, stream>>>(out_proj_w, wout_bf, n2);
    cvt_bf16_kernel<<<(n3 + 255) / 256, 256, 0, stream>>>(mlp_w1, w1_bf, n3);
    cvt_bf16_kernel<<<(n3 + 255) / 256, 256, 0, stream>>>(mlp_w2, w2_bf, n3);
  }
  hipMemsetAsync(zbuf, 0, TOKC * 4, stream);

  const dim3 gemmBlk(256);
  for (int it = 0; it < NUM_ITERS; ++it) {
    const float* emb = iter_emb + (size_t)imin_host(it, 30) * C_SZ;

    // LN1 on z_ctx = z + 0.1*emb + u
    ln_kernel<1><<<NTOK / 8, 256, 0, stream>>>(zbuf, u, emb, ln1_w, ln1_b, x_bf);

    // QKV GEMM [4096,768]x[768,2304] + pack heads (q pre-scaled)
    gemm_bf16_kernel<1><<<dim3(2304 / 128, NTOK / 128), gemmBlk, 0, stream>>>(
        x_bf, wqkv_bf, in_proj_b, nullptr, nullptr, nullptr, nullptr, qb, kb, vb,
        NTOK, 3 * C_SZ, C_SZ);

    // attention
    flash_attn_kernel<<<dim3(T_SZ / 64, B_SZ * H_SZ), 128, 0, stream>>>(qb, kb, vb,
                                                                        attn_bf);

    // out-proj GEMM + bias + residual(z) -> z_attn
    gemm_bf16_kernel<2><<<dim3(C_SZ / 128, NTOK / 128), gemmBlk, 0, stream>>>(
        attn_bf, wout_bf, out_proj_b, z_attn, nullptr, zbuf, nullptr, nullptr,
        nullptr, nullptr, NTOK, C_SZ, C_SZ);

    // LN2
    ln_kernel<0><<<NTOK / 8, 256, 0, stream>>>(z_attn, nullptr, nullptr, ln2_w,
                                               ln2_b, h_bf);

    // MLP fc1 + exact GELU -> bf16
    gemm_bf16_kernel<3><<<dim3(4 * C_SZ / 128, NTOK / 128), gemmBlk, 0, stream>>>(
        h_bf, w1_bf, mlp_b1, nullptr, g_bf, nullptr, nullptr, nullptr, nullptr,
        nullptr, NTOK, 4 * C_SZ, C_SZ);

    // MLP fc2 + bias + z_attn - z -> delta, written directly into Fs ring slot
    const int newest = it % M_HIST;
    float* FsSlot = FsBuf + (size_t)newest * TOKC;
    gemm_bf16_kernel<4><<<dim3(C_SZ / 128, NTOK / 128), gemmBlk, 0, stream>>>(
        g_bf, w2_bf, mlp_b2, FsSlot, nullptr, z_attn, zbuf, nullptr, nullptr,
        nullptr, NTOK, C_SZ, 4 * C_SZ);

    // Anderson update
    const int K = imin_host(it, M_HIST - 1);
    int olds[4] = {0, 0, 0, 0};
    for (int j = 0; j < K; ++j) olds[j] = (it - K + j) % M_HIST;
    anderson_kernel<<<NTOK / 8, 256, 0, stream>>>(zbuf, FsBuf, newest, K, olds[0],
                                                  olds[1], olds[2], olds[3]);
  }

  hipMemcpyAsync(d_out, zbuf, TOKC * 4, hipMemcpyDeviceToDevice, stream);
}
